// GNNModel_82403242541085
// MI455X (gfx1250) — matrix operand order repaired and checked by reading, not compiled, not run on hardware
//
#include <hip/hip_runtime.h>
#include <hip/hip_bf16.h>

#define NPAT 100000
#define NDIS 30000
#define NMED 20000
#define EDG  600000
#define ELS  200000
#define HD   128
#define CD   64

typedef __attribute__((ext_vector_type(16))) __bf16 v16bf;
typedef __attribute__((ext_vector_type(8)))  float  v8f;

__device__ __forceinline__ unsigned short f2bf(float f) {
    unsigned int u = __float_as_uint(f);
    u += 0x7FFFu + ((u >> 16) & 1u);           // round-to-nearest-even
    return (unsigned short)(u >> 16);
}

// ---------------- utility kernels ----------------
__global__ void k_fill(float* p, long n, float v) {
    long i = blockIdx.x * (long)blockDim.x + threadIdx.x;
    if (i < n) p[i] = v;
}

__global__ void k_cast_bf16(const float* __restrict__ x, unsigned short* __restrict__ y, long n) {
    long i = blockIdx.x * (long)blockDim.x + threadIdx.x;
    if (i < n) y[i] = f2bf(x[i]);
}

// W[k, c] (k<128, c<Cout)  ->  Wt[c*128 + k]  (bf16, B-operand friendly: contiguous K per column)
__global__ void k_w_transpose(const float* __restrict__ W, unsigned short* __restrict__ Wt, int Cout) {
    int c = blockIdx.x;        // 0..Cout-1
    int k = threadIdx.x;       // 0..127
    Wt[c * HD + k] = f2bf(W[k * Cout + c]);
}

// v[k] = sum_c W[k,c] * a[c]   (folds W_dst @ a_dst so hd never needs materializing)
__global__ void k_w_dot_a(const float* __restrict__ W, const float* __restrict__ a,
                          float* __restrict__ v, int Cout) {
    int k = threadIdx.x;       // 128 threads
    float s = 0.f;
    for (int c = 0; c < Cout; ++c) s += W[k * Cout + c] * a[c];
    v[k] = s;
}

// alpha[n] = dot(X[n,:128], v)
__global__ void k_alpha(const float* __restrict__ X, const float* __restrict__ v,
                        float* __restrict__ alpha, int N) {
    int n = blockIdx.x * blockDim.x + threadIdx.x;
    if (n >= N) return;
    const float4* xr = (const float4*)(X + (long)n * HD);
    float s = 0.f;
    #pragma unroll 8
    for (int q = 0; q < HD / 4; ++q) {
        float4 t = xr[q];
        s += t.x * v[4*q] + t.y * v[4*q+1] + t.z * v[4*q+2] + t.w * v[4*q+3];
    }
    alpha[n] = s;
}

// ---------------- WMMA GEMM: Y[N,COUT] = Xb[N,128](bf16) @ W(bf16, pre-transposed) ----------------
// COUT is a template constant (128 or 64) so the D-matrix writeback folds the row stride into
// immediate store offsets (one address + 8-store clause) instead of chained 64-bit adds.
template <int COUT>
__global__ void k_wmma_gemm(const unsigned short* __restrict__ Xb,  // [N,128] bf16 row-major
                            const unsigned short* __restrict__ Wt,  // [COUT,128] bf16 (K contiguous)
                            float* __restrict__ Y,                  // [N,COUT] f32
                            int N) {
    const int lane    = threadIdx.x & 31;
    const int wid     = threadIdx.x >> 5;          // 8 waves / block
    const int rowTile = blockIdx.x * 8 + wid;
    const int colTile = blockIdx.y;
    if (rowTile * 16 >= N) return;                 // wave-uniform branch: EXEC stays all-1 for WMMA
    const int half = lane >> 4;
    const int lr   = lane & 15;
    const long arow = (long)(rowTile * 16 + lr) * HD;
    const long brow = (long)(colTile * 16 + lr) * HD;

    v8f acc = {0.f, 0.f, 0.f, 0.f, 0.f, 0.f, 0.f, 0.f};
    #pragma unroll
    for (int ks = 0; ks < HD / 32; ++ks) {
        union { v16bf v; uint4 q[2]; } A, B;
        // A 16x32 bf16 layout: lane<16 row=lr holds K {0..7,16..23}; lane>=16 holds {8..15,24..31}
        const unsigned short* ap = Xb + arow + ks * 32 + half * 8;
        A.q[0] = *(const uint4*)(ap);
        A.q[1] = *(const uint4*)(ap + 16);
        // B 32x16 bf16 layout: col=lr, lanes 0-15 K {0..7,16..23}, lanes 16-31 K {8..15,24..31}
        const unsigned short* bp = Wt + brow + ks * 32 + half * 8;
        B.q[0] = *(const uint4*)(bp);
        B.q[1] = *(const uint4*)(bp + 16);
        acc = __builtin_amdgcn_wmma_f32_16x16x32_bf16(false, A.v, false, B.v,
                                                      (short)0, acc, false, false);
    }
    // C/D layout: VGPR r -> row (r + half*8), col = lr.  COUT constant -> immediate offsets.
    float* yp = Y + (long)(rowTile * 16 + half * 8) * COUT + colTile * 16 + lr;
    #pragma unroll
    for (int r = 0; r < 8; ++r)
        yp[(long)r * COUT] = acc[r];
}

// ---------------- edge-phase kernels ----------------
__device__ __forceinline__ void atomicMaxF(float* addr, float val) {
    int* ai = (int*)addr;
    int cur = __float_as_int(*addr);
    while (val > __int_as_float(cur)) {
        int old = atomicCAS(ai, cur, __float_as_int(val));
        if (old == cur) break;
        cur = old;
    }
}

__global__ void k_edge_logits(const int* __restrict__ src, const int* __restrict__ dst,
                              const float* __restrict__ as_, const float* __restrict__ ad_,
                              float* __restrict__ ebuf, float* __restrict__ mbuf, int ne) {
    int e = blockIdx.x * blockDim.x + threadIdx.x;
    if (e >= ne) return;
    float x = as_[src[e]] + ad_[dst[e]];
    x = (x >= 0.f) ? x : 0.2f * x;                 // leaky_relu(0.2)
    ebuf[e] = x;
    atomicMaxF(&mbuf[dst[e]], x);
}

__global__ void k_edge_exp(const int* __restrict__ dst, const float* __restrict__ ebuf,
                           const float* __restrict__ mbuf, float* __restrict__ exbuf,
                           float* __restrict__ dbuf, int ne) {
    int e = blockIdx.x * blockDim.x + threadIdx.x;
    if (e >= ne) return;
    int d = dst[e];
    float ex = __expf(ebuf[e] - mbuf[d]);
    exbuf[e] = ex;
    atomicAdd(&dbuf[d], ex);
}

// one wave per edge: out[dst] += (ex/denom) * hs[src]
__global__ void k_edge_scatter(const int* __restrict__ src, const int* __restrict__ dst,
                               const float* __restrict__ exbuf, const float* __restrict__ dbuf,
                               const float* __restrict__ hs, float* __restrict__ out,
                               int ne, int K) {
    int e = blockIdx.x * 8 + (threadIdx.x >> 5);
    int lane = threadIdx.x & 31;
    if (e >= ne) return;
    int s = src[e], d = dst[e];
    float w = exbuf[e] / dbuf[d];
    const float* hrow = hs + (long)s * K;
    float* orow = out + (long)d * K;
    for (int f = lane; f < K; f += 32)
        atomicAdd(&orow[f], w * hrow[f]);
}

__global__ void k_bias_act(float* __restrict__ y, const float* __restrict__ b0,
                           const float* __restrict__ b1p, long n, int K, int relu) {
    long i = blockIdx.x * (long)blockDim.x + threadIdx.x;
    if (i >= n) return;
    int f = (int)(i % K);
    float v = y[i] + b0[f];
    if (b1p) v += b1p[f];
    if (relu) v = fmaxf(v, 0.f);
    y[i] = v;
}

__global__ void k_predict(const int* __restrict__ row, const int* __restrict__ col,
                          const float* __restrict__ zp, const float* __restrict__ zm,
                          const float* __restrict__ w, const float* __restrict__ b,
                          float* __restrict__ out, int ne) {
    int e = blockIdx.x * blockDim.x + threadIdx.x;
    if (e >= ne) return;
    const float* p = zp + (long)row[e] * CD;
    const float* m = zm + (long)col[e] * CD;
    float s = b[0];
    #pragma unroll 8
    for (int c = 0; c < CD; ++c) s += p[c] * w[c] + m[c] * w[CD + c];
    out[e] = s;
}

// ---------------- host orchestration ----------------
extern "C" void kernel_launch(void* const* d_in, const int* in_sizes, int n_in,
                              void* d_out, int out_size, void* d_ws, size_t ws_size,
                              hipStream_t stream) {
    (void)in_sizes; (void)n_in; (void)out_size; (void)ws_size;

    const int* SRC[4] = {(const int*)d_in[0], (const int*)d_in[2], (const int*)d_in[4], (const int*)d_in[6]};
    const int* DST[4] = {(const int*)d_in[1], (const int*)d_in[3], (const int*)d_in[5], (const int*)d_in[7]};
    const int  st[4]  = {0, 1, 0, 2};   // src node type per edge type (0=pat,1=dis,2=med)
    const int  dt[4]  = {1, 0, 2, 0};   // dst node type
    const int  NN[3]  = {NPAT, NDIS, NMED};
    const int* rowI = (const int*)d_in[8];
    const int* colI = (const int*)d_in[9];
    const float* emb[3] = {(const float*)d_in[10], (const float*)d_in[11], (const float*)d_in[12]};
    const float* W1s = (const float*)d_in[13]; const float* W1d = (const float*)d_in[14];
    const float* a1s = (const float*)d_in[15]; const float* a1d = (const float*)d_in[16];
    const float* b1  = (const float*)d_in[17];
    const float* W2s = (const float*)d_in[18]; const float* W2d = (const float*)d_in[19];
    const float* a2s = (const float*)d_in[20]; const float* a2d = (const float*)d_in[21];
    const float* b2  = (const float*)d_in[22];
    const float* linw = (const float*)d_in[23]; const float* linb = (const float*)d_in[24];

    char* ws = (char*)d_ws; size_t off = 0;
    auto alloc = [&](size_t bytes) -> void* {
        void* p = ws + off; off = (off + bytes + 255) & ~(size_t)255; return p;
    };

    unsigned short* xb[3];  for (int t = 0; t < 3; ++t) xb[t] = (unsigned short*)alloc((size_t)NN[t] * HD * 2);
    unsigned short* hb[3];  for (int t = 0; t < 3; ++t) hb[t] = (unsigned short*)alloc((size_t)NN[t] * HD * 2);
    float* o1[3];           for (int t = 0; t < 3; ++t) o1[t] = (float*)alloc((size_t)NN[t] * HD * 4);
    float* o2[3];           for (int t = 0; t < 3; ++t) o2[t] = (float*)alloc((size_t)NN[t] * CD * 4);
    unsigned short* W1t[4]; for (int i = 0; i < 4; ++i) W1t[i] = (unsigned short*)alloc((size_t)HD * HD * 2);
    unsigned short* W2t[4]; for (int i = 0; i < 4; ++i) W2t[i] = (unsigned short*)alloc((size_t)CD * HD * 2);
    float* v1sv[4]; float* v1dv[4]; float* v2sv[4]; float* v2dv[4];
    for (int i = 0; i < 4; ++i) {
        v1sv[i] = (float*)alloc(HD * 4); v1dv[i] = (float*)alloc(HD * 4);
        v2sv[i] = (float*)alloc(HD * 4); v2dv[i] = (float*)alloc(HD * 4);
    }
    float* as_   = (float*)alloc((size_t)NPAT * 4);
    float* ad_   = (float*)alloc((size_t)NPAT * 4);
    float* hsbuf = (float*)alloc((size_t)NPAT * HD * 4);   // reused per edge type (layer2 fits)
    float* ebuf  = (float*)alloc((size_t)EDG * 4);
    float* exbuf = (float*)alloc((size_t)EDG * 4);
    float* mbuf  = (float*)alloc((size_t)NPAT * 4);
    float* dbuf  = (float*)alloc((size_t)NPAT * 4);

    const int TB = 256;
    auto nb = [&](long n) { return (unsigned)((n + TB - 1) / TB); };
    const float NEG_INF = -__builtin_inff();

    // casts + weight prep
    for (int t = 0; t < 3; ++t)
        k_cast_bf16<<<nb((long)NN[t] * HD), TB, 0, stream>>>(emb[t], xb[t], (long)NN[t] * HD);
    for (int i = 0; i < 4; ++i) {
        k_w_transpose<<<HD, HD, 0, stream>>>(W1s + (size_t)i * HD * HD, W1t[i], HD);
        k_w_transpose<<<CD, HD, 0, stream>>>(W2s + (size_t)i * HD * CD, W2t[i], CD);
        k_w_dot_a<<<1, HD, 0, stream>>>(W1s + (size_t)i * HD * HD, a1s + (size_t)i * HD, v1sv[i], HD);
        k_w_dot_a<<<1, HD, 0, stream>>>(W1d + (size_t)i * HD * HD, a1d + (size_t)i * HD, v1dv[i], HD);
        k_w_dot_a<<<1, HD, 0, stream>>>(W2s + (size_t)i * HD * CD, a2s + (size_t)i * CD, v2sv[i], CD);
        k_w_dot_a<<<1, HD, 0, stream>>>(W2d + (size_t)i * HD * CD, a2d + (size_t)i * CD, v2dv[i], CD);
    }

    // ---------- layer 1 (H -> H) ----------
    for (int t = 0; t < 3; ++t)
        k_fill<<<nb((long)NN[t] * HD), TB, 0, stream>>>(o1[t], (long)NN[t] * HD, 0.f);
    for (int i = 0; i < 4; ++i) {
        int s = st[i], d = dt[i], Ns = NN[s], Nd = NN[d];
        dim3 gg((Ns / 16 + 7) / 8, HD / 16);
        k_wmma_gemm<HD><<<gg, 256, 0, stream>>>(xb[s], W1t[i], hsbuf, Ns);
        k_alpha<<<nb(Ns), TB, 0, stream>>>(emb[s], v1sv[i], as_, Ns);
        k_alpha<<<nb(Nd), TB, 0, stream>>>(emb[d], v1dv[i], ad_, Nd);
        k_fill<<<nb(Nd), TB, 0, stream>>>(mbuf, Nd, NEG_INF);
        k_fill<<<nb(Nd), TB, 0, stream>>>(dbuf, Nd, 0.f);
        k_edge_logits<<<nb(EDG), TB, 0, stream>>>(SRC[i], DST[i], as_, ad_, ebuf, mbuf, EDG);
        k_edge_exp<<<nb(EDG), TB, 0, stream>>>(DST[i], ebuf, mbuf, exbuf, dbuf, EDG);
        k_edge_scatter<<<(EDG + 7) / 8, 256, 0, stream>>>(SRC[i], DST[i], exbuf, dbuf, hsbuf, o1[d], EDG, HD);
    }
    k_bias_act<<<nb((long)NPAT * HD), TB, 0, stream>>>(o1[0], b1 + 1 * HD, b1 + 3 * HD, (long)NPAT * HD, HD, 1);
    k_bias_act<<<nb((long)NDIS * HD), TB, 0, stream>>>(o1[1], b1 + 0 * HD, nullptr,     (long)NDIS * HD, HD, 1);
    k_bias_act<<<nb((long)NMED * HD), TB, 0, stream>>>(o1[2], b1 + 2 * HD, nullptr,     (long)NMED * HD, HD, 1);
    for (int t = 0; t < 3; ++t)
        k_cast_bf16<<<nb((long)NN[t] * HD), TB, 0, stream>>>(o1[t], hb[t], (long)NN[t] * HD);

    // ---------- layer 2 (H -> C) ----------
    for (int t = 0; t < 3; ++t)
        k_fill<<<nb((long)NN[t] * CD), TB, 0, stream>>>(o2[t], (long)NN[t] * CD, 0.f);
    for (int i = 0; i < 4; ++i) {
        int s = st[i], d = dt[i], Ns = NN[s], Nd = NN[d];
        dim3 gg((Ns / 16 + 7) / 8, CD / 16);
        k_wmma_gemm<CD><<<gg, 256, 0, stream>>>(hb[s], W2t[i], hsbuf, Ns);
        k_alpha<<<nb(Ns), TB, 0, stream>>>(o1[s], v2sv[i], as_, Ns);
        k_alpha<<<nb(Nd), TB, 0, stream>>>(o1[d], v2dv[i], ad_, Nd);
        k_fill<<<nb(Nd), TB, 0, stream>>>(mbuf, Nd, NEG_INF);
        k_fill<<<nb(Nd), TB, 0, stream>>>(dbuf, Nd, 0.f);
        k_edge_logits<<<nb(EDG), TB, 0, stream>>>(SRC[i], DST[i], as_, ad_, ebuf, mbuf, EDG);
        k_edge_exp<<<nb(EDG), TB, 0, stream>>>(DST[i], ebuf, mbuf, exbuf, dbuf, EDG);
        k_edge_scatter<<<(EDG + 7) / 8, 256, 0, stream>>>(SRC[i], DST[i], exbuf, dbuf, hsbuf, o2[d], EDG, CD);
    }
    k_bias_act<<<nb((long)NPAT * CD), TB, 0, stream>>>(o2[0], b2 + 1 * CD, b2 + 3 * CD, (long)NPAT * CD, CD, 0);
    k_bias_act<<<nb((long)NDIS * CD), TB, 0, stream>>>(o2[1], b2 + 0 * CD, nullptr,     (long)NDIS * CD, CD, 0);
    k_bias_act<<<nb((long)NMED * CD), TB, 0, stream>>>(o2[2], b2 + 2 * CD, nullptr,     (long)NMED * CD, CD, 0);

    // ---------- link predictor ----------
    k_predict<<<nb(ELS), TB, 0, stream>>>(rowI, colI, o2[0], o2[2], linw, linb, (float*)d_out, ELS);
}
